// Cluster_GT_33088428048634
// MI455X (gfx1250) — compile-verified, matrix operand order checked by reading, not run on hardware
//
#include <hip/hip_runtime.h>
#include <hip/hip_bf16.h>

// ---------------------------------------------------------------------------
// Types for WMMA fragments / TDM descriptors (wave32, gfx1250)
// ---------------------------------------------------------------------------
typedef __attribute__((ext_vector_type(16))) _Float16 v16h;
typedef __attribute__((ext_vector_type(8)))  _Float16 v8h;
typedef __attribute__((ext_vector_type(8)))  float    v8f;
typedef __attribute__((ext_vector_type(4)))  unsigned int v4u;
typedef __attribute__((ext_vector_type(4)))  int      v4i;
typedef __attribute__((ext_vector_type(8)))  int      v8i;

#define WMMA_F16(a, b, c) \
  __builtin_amdgcn_wmma_f32_16x16x32_f16(false, (a), false, (b), (short)0, (c), false, false)

// Dimensions (fixed by the reference)
#define NN 32768
#define SS 32768
#define CC 1024
#define EE 16384
#define DD 64
#define BB 16
#define PP 64
#define NCC 10

// Load a 16x32 f16 WMMA operand fragment from a row-major [16, ld] tile.
// Layout (ISA 7.12.2, 16-bit A 16x32): lanes 0-15 -> rows 0-15, K {0..7,16..23};
// lanes 16-31 -> rows 0-15, K {8..15,24..31}.  Two 16B loads per lane.
__device__ __forceinline__ v16h load_frag16(const _Float16* base, int ld, int lane) {
  int r  = lane & 15;
  int h8 = (lane >> 4) << 3;             // 0 or 8
  const _Float16* p = base + r * ld + h8;
  v8h lo = *(const v8h*)(p);
  v8h hi = *(const v8h*)(p + 16);
  return __builtin_shufflevector(lo, hi, 0,1,2,3,4,5,6,7,8,9,10,11,12,13,14,15);
}

// ---------------------------------------------------------------------------
// Tensor Data Mover: DMA a contiguous run of `elems` f16 values from global
// memory into LDS at byte offset `lds_off` (one 1 x elems 2-byte tile).
// D# layout per ISA cdna5 ch.8: group0 {count, lds_addr, global_addr, type=2},
// group1 {data_size=2B, tensor_dim0/1, tile_dim0/1, stride}.
// Issue from ONE wave only (TDM ignores EXEC; every executing wave issues).
// ---------------------------------------------------------------------------
__device__ __forceinline__ void tdm_load_1d(const void* gptr, unsigned lds_off,
                                            unsigned elems) {
  unsigned long long ga = (unsigned long long)(uintptr_t)gptr;
  v4u g0;
  g0[0] = 1u;                                          // count=1 (valid user D#)
  g0[1] = lds_off;                                     // lds_addr (bytes)
  g0[2] = (unsigned)(ga & 0xffffffffu);                // global_addr[31:0]
  g0[3] = (unsigned)((ga >> 32) & 0x01ffffffu) | (2u << 30);  // addr[56:32] | type=2
  v8i g1;
  g1[0] = (int)(1u << 16);                             // wg_mask=0, data_size=1 (2B)
  g1[1] = (int)((elems & 0xffffu) << 16);              // tensor_dim0[15:0]
  g1[2] = (int)(((elems >> 16) & 0xffffu) | (1u << 16));  // dim0[31:16] | tensor_dim1=1
  g1[3] = (int)((elems & 0xffffu) << 16);              // dim1 hi=0 | tile_dim0=elems
  g1[4] = 1;                                           // tile_dim1=1, tile_dim2=0
  g1[5] = (int)elems;                                  // tensor_dim0_stride[31:0]
  g1[6] = 0;
  g1[7] = 0;
  v4i z4 = {0, 0, 0, 0};
#if __clang_major__ >= 23
  v8i z8 = {0, 0, 0, 0, 0, 0, 0, 0};
  __builtin_amdgcn_tensor_load_to_lds(g0, g1, z4, z4, z8, 0);
#else
  __builtin_amdgcn_tensor_load_to_lds(g0, g1, z4, z4, 0);
#endif
}

__device__ __forceinline__ unsigned lds_offset_of(const void* p) {
  // LDS flat addresses carry the wave-relative LDS byte offset in addr[31:0]
  return (unsigned)(uintptr_t)p;
}

// ---------------------------------------------------------------------------
// Fused two-layer MLP:  out = act2( relu(in@W1+b1) @ W2 + b2 )
// Weights pre-transposed f16: Wt[n][k], DMA'd once per block into LDS by the
// Tensor Data Mover.  Block = 256 threads = 8 waves; each wave computes a
// 16x64 output tile; block covers 128 rows.  Layer-1 results pass through a
// per-wave LDS tile (C-layout -> A-layout transpose).
// ---------------------------------------------------------------------------
__global__ void __launch_bounds__(256) k_mlp2(
    const _Float16* __restrict__ in16, const _Float16* __restrict__ W1t,
    const float* __restrict__ b1, const _Float16* __restrict__ W2t,
    const float* __restrict__ b2, float* __restrict__ out32,
    _Float16* __restrict__ out16, int relu2)
{
  __shared__ _Float16 wlds[2][4096];   // staged W1t, W2t
  __shared__ _Float16 tile[8][16 * 64];
  int lane = threadIdx.x & 31, wave = threadIdx.x >> 5;
  int row0 = blockIdx.x * 128 + wave * 16;
  const _Float16* A = in16 + (size_t)row0 * 64;
  int cn = lane & 15, rhi = (lane >> 4) * 8;

  if (threadIdx.x < 32) {              // wave 0 only: issue TDM DMAs
    tdm_load_1d(W1t, lds_offset_of(&wlds[0][0]), 4096);
    tdm_load_1d(W2t, lds_offset_of(&wlds[1][0]), 4096);
    __builtin_amdgcn_s_wait_tensorcnt(0);
  }
  __syncthreads();

  v8f acc[4] = {};
  for (int kc = 0; kc < 64; kc += 32) {
    v16h af = load_frag16(A + kc, 64, lane);
#pragma unroll
    for (int nt = 0; nt < 4; ++nt) {
      v16h bf = load_frag16(&wlds[0][0] + nt * 16 * 64 + kc, 64, lane);
      acc[nt] = WMMA_F16(af, bf, acc[nt]);
    }
  }
#pragma unroll
  for (int nt = 0; nt < 4; ++nt) {
    int col = nt * 16 + cn;
    float bb = b1[col];
#pragma unroll
    for (int i = 0; i < 8; ++i) {
      float v = acc[nt][i] + bb;
      v = v > 0.f ? v : 0.f;
      tile[wave][(i + rhi) * 64 + col] = (_Float16)v;   // C-layout -> row-major
    }
  }
  __syncthreads();

  v8f acc2[4] = {};
  for (int kc = 0; kc < 64; kc += 32) {
    v16h af = load_frag16(&tile[wave][0] + kc, 64, lane);
#pragma unroll
    for (int nt = 0; nt < 4; ++nt) {
      v16h bf = load_frag16(&wlds[1][0] + nt * 16 * 64 + kc, 64, lane);
      acc2[nt] = WMMA_F16(af, bf, acc2[nt]);
    }
  }
#pragma unroll
  for (int nt = 0; nt < 4; ++nt) {
    int col = nt * 16 + cn;
    float bb = b2[col];
#pragma unroll
    for (int i = 0; i < 8; ++i) {
      float v = acc2[nt][i] + bb;
      if (relu2) v = v > 0.f ? v : 0.f;
      size_t o = (size_t)(row0 + rhi + i) * 64 + col;
      if (out32) out32[o] = v;
      if (out16) out16[o] = (_Float16)v;
    }
  }
}

// Single-layer GEMM:  out16 = in@W + b   (no activation), W via TDM into LDS.
__global__ void __launch_bounds__(256) k_gemm1(
    const _Float16* __restrict__ in16, const _Float16* __restrict__ Wt,
    const float* __restrict__ bias, _Float16* __restrict__ out16)
{
  __shared__ _Float16 wlds[4096];
  int lane = threadIdx.x & 31, wave = threadIdx.x >> 5;
  int row0 = blockIdx.x * 128 + wave * 16;
  const _Float16* A = in16 + (size_t)row0 * 64;

  if (threadIdx.x < 32) {
    tdm_load_1d(Wt, lds_offset_of(&wlds[0]), 4096);
    __builtin_amdgcn_s_wait_tensorcnt(0);
  }
  __syncthreads();

  v8f acc[4] = {};
  for (int kc = 0; kc < 64; kc += 32) {
    v16h af = load_frag16(A + kc, 64, lane);
#pragma unroll
    for (int nt = 0; nt < 4; ++nt) {
      v16h bf = load_frag16(&wlds[0] + nt * 16 * 64 + kc, 64, lane);
      acc[nt] = WMMA_F16(af, bf, acc[nt]);
    }
  }
  int cn = lane & 15, rhi = (lane >> 4) * 8;
#pragma unroll
  for (int nt = 0; nt < 4; ++nt) {
    int col = nt * 16 + cn;
    float bb = bias[col];
#pragma unroll
    for (int i = 0; i < 8; ++i)
      out16[(size_t)(row0 + rhi + i) * 64 + col] = (_Float16)(acc[nt][i] + bb);
  }
}

// ---------------------------------------------------------------------------
// Per-cluster outer products: Ms[c] = sum_{s in c} ck[s] (x) V[s]   (128x64)
// One block (8 waves) per cluster; chunks of 32 entries staged transposed into
// LDS so both fragments are contiguous 16B ds loads; one WMMA per (tile,chunk).
// ---------------------------------------------------------------------------
__global__ void __launch_bounds__(256) k_ms(
    const _Float16* __restrict__ ck16, const _Float16* __restrict__ V16,
    const int* __restrict__ offs, const int* __restrict__ cnts,
    float* __restrict__ Ms)
{
  __shared__ _Float16 ckT[128 * 32];   // ckT[i][k]
  __shared__ _Float16 vT[64 * 32];     // vT[j][k]
  int c = blockIdx.x;
  int tid = threadIdx.x, lane = tid & 31, wave = tid >> 5;
  int s0 = offs[c], n = cnts[c];

  v8f acc[4] = {};
  for (int base = 0; base < n; base += 32) {
    int valid = n - base; if (valid > 32) valid = 32;
    for (int t = tid; t < 4096; t += 256) {
      int sl = t >> 7, i = t & 127;
      _Float16 v = (_Float16)0.f;
      if (sl < valid) v = ck16[(size_t)(s0 + base + sl) * 128 + i];
      ckT[i * 32 + sl] = v;
    }
    for (int t = tid; t < 2048; t += 256) {
      int sl = t >> 6, j = t & 63;
      _Float16 v = (_Float16)0.f;
      if (sl < valid) v = V16[(size_t)(s0 + base + sl) * 64 + j];
      vT[j * 32 + sl] = v;
    }
    __syncthreads();
    v16h af = load_frag16(&ckT[wave * 16 * 32], 32, lane);
#pragma unroll
    for (int nt = 0; nt < 4; ++nt) {
      v16h bf = load_frag16(&vT[nt * 16 * 32], 32, lane);
      acc[nt] = WMMA_F16(af, bf, acc[nt]);
    }
    __syncthreads();
  }
  int cn = lane & 15, rhi = (lane >> 4) * 8;
#pragma unroll
  for (int nt = 0; nt < 4; ++nt)
#pragma unroll
    for (int i = 0; i < 8; ++i) {
      int ii = wave * 16 + rhi + i;
      int j = nt * 16 + cn;
      Ms[((size_t)c * 128 + ii) * 64 + j] = acc[nt][i];
    }
}

// ---------------------------------------------------------------------------
// Small helper kernels
// ---------------------------------------------------------------------------
__global__ void k_cvt(const float* __restrict__ s, _Float16* __restrict__ d, int n) {
  int i = blockIdx.x * blockDim.x + threadIdx.x;
  if (i < n) d[i] = (_Float16)s[i];
}

__global__ void k_wt(const float* __restrict__ W, _Float16* __restrict__ Wt) { // 64x64 -> transposed f16
  int t = blockIdx.x * blockDim.x + threadIdx.x;
  if (t < 4096) { int k = t >> 6, n = t & 63; Wt[n * 64 + k] = (_Float16)W[t]; }
}

__global__ void k_gather(const _Float16* __restrict__ h16, const int* __restrict__ map,
                         _Float16* __restrict__ sx16) {
  int idx = blockIdx.x * blockDim.x + threadIdx.x;
  if (idx >= SS * 8) return;
  int s = idx >> 3, q = idx & 7;
  const uint4* src = (const uint4*)(h16 + (size_t)map[s] * 64);
  uint4* dst = (uint4*)(sx16 + (size_t)s * 64);
  dst[q] = src[q];
}

__global__ void k_count(const int* __restrict__ b, int* __restrict__ cnts) {
  int s = blockIdx.x * blockDim.x + threadIdx.x;
  if (s < SS) atomicAdd(&cnts[b[s]], 1);
}

__global__ void __launch_bounds__(1024) k_scan(const int* __restrict__ cnts, int* __restrict__ offs) {
  __shared__ int sm[1024];
  int t = threadIdx.x;
  int v = cnts[t];
  sm[t] = v;
  __syncthreads();
  for (int d = 1; d < 1024; d <<= 1) {
    int x = (t >= d) ? sm[t - d] : 0;
    __syncthreads();
    sm[t] += x;
    __syncthreads();
  }
  offs[t] = sm[t] - v;   // exclusive prefix
}

__global__ void k_deg(const int* __restrict__ row, const float* __restrict__ attr,
                      float* __restrict__ deg) {
  int e = blockIdx.x * blockDim.x + threadIdx.x;
  if (e < EE) atomicAdd(&deg[row[e]], attr[e]);
}

__global__ void k_ea(const int* __restrict__ row, const int* __restrict__ col,
                     const float* __restrict__ attr, const float* __restrict__ deg,
                     float* __restrict__ ea) {
  int e = blockIdx.x * blockDim.x + threadIdx.x;
  if (e >= EE) return;
  float dr = deg[row[e]], dc = deg[col[e]];
  float ir = dr > 0.f ? rsqrtf(dr) : 0.f;
  float ic = dc > 0.f ? rsqrtf(dc) : 0.f;
  ea[e] = ir * attr[e] * ic;
}

__global__ void k_segsumQK(const int* __restrict__ b, const _Float16* __restrict__ Q16,
                           const _Float16* __restrict__ K16, float* __restrict__ Qs,
                           float* __restrict__ Ksum) {
  int idx = blockIdx.x * blockDim.x + threadIdx.x;
  if (idx >= SS * 64) return;
  int s = idx >> 6, d = idx & 63, c = b[s];
  atomicAdd(&Qs[c * 64 + d], (float)Q16[idx]);
  atomicAdd(&Ksum[c * 64 + d], (float)K16[idx]);
}

__global__ void k_alpha(const float* __restrict__ ab, float* __restrict__ sab) {
  float a0 = ab[0], a1 = ab[1];
  float m = a0 > a1 ? a0 : a1;
  float e0 = __expf(a0 - m), e1 = __expf(a1 - m);
  float s = e0 + e1;
  sab[0] = rsqrtf(e0 / s);
  sab[1] = rsqrtf(e1 / s);
}

__global__ void k_cqKs(const float* __restrict__ Ksum, const int* __restrict__ cnts,
                       const float* __restrict__ Qs2, const float* __restrict__ sab,
                       float* __restrict__ Ksk, float* __restrict__ cq) {
  int idx = blockIdx.x * blockDim.x + threadIdx.x;
  if (idx >= CC * 64) return;
  int c = idx >> 6, d = idx & 63;
  float cnt = (float)(cnts[c] > 1 ? cnts[c] : 1);
  float ks = Ksum[idx] / cnt;
  Ksk[idx] = ks > 0.f ? ks : 0.f;
  float qk = Qs2[idx] > 0.f ? Qs2[idx] : 0.f;
  cq[c * 128 + d]      = sab[0] * qk;
  cq[c * 128 + 64 + d] = sab[1] * qk;
}

__global__ void k_ck(const int* __restrict__ b, const _Float16* __restrict__ K16,
                     const float* __restrict__ Ksk, const float* __restrict__ sab,
                     _Float16* __restrict__ ck16, float* __restrict__ ck_s) {
  int idx = blockIdx.x * blockDim.x + threadIdx.x;
  if (idx >= SS * 64) return;
  int s = idx >> 6, d = idx & 63, c = b[s];
  float kk = (float)K16[idx]; kk = kk > 0.f ? kk : 0.f;
  float v0 = sab[0] * Ksk[c * 64 + d];
  float v1 = sab[1] * kk;
  ck16[(size_t)s * 128 + d]      = (_Float16)v0;
  ck16[(size_t)s * 128 + 64 + d] = (_Float16)v1;
  atomicAdd(&ck_s[c * 128 + d], v0);
  atomicAdd(&ck_s[c * 128 + 64 + d], v1);
}

// Edge propagation on coarsened graph: Ms2[col] += ea*Ms[row]; cks2 likewise.
// Ms (33MB) is L2-resident; atomics land in L2.  Prefetch the source row.
__global__ void __launch_bounds__(256) k_prop(
    const int* __restrict__ row, const int* __restrict__ col,
    const float* __restrict__ ea, const float* __restrict__ Ms,
    const float* __restrict__ ck_s, float* __restrict__ Ms2,
    float* __restrict__ cks2)
{
  int e = blockIdx.x;
  float a = ea[e];
  int r = row[e], cl = col[e];
  const float* src = Ms + (size_t)r * 8192;
  float* dst = Ms2 + (size_t)cl * 8192;
  __builtin_prefetch(&src[threadIdx.x * 32], 0, 1);   // global_prefetch_b8
  for (int i = threadIdx.x; i < 8192; i += 256)
    atomicAdd(&dst[i], a * src[i]);
  if (threadIdx.x < 128)
    atomicAdd(&cks2[cl * 128 + threadIdx.x], a * ck_s[r * 128 + threadIdx.x]);
}

__global__ void __launch_bounds__(128) k_hout(
    const float* __restrict__ cq, const float* __restrict__ cks2,
    const float* __restrict__ Ms2, float* __restrict__ outc)
{
  __shared__ float red[128];
  int c = blockIdx.x, t = threadIdx.x;
  red[t] = cq[c * 128 + t] * cks2[c * 128 + t];
  __syncthreads();
  for (int s = 64; s > 0; s >>= 1) {
    if (t < s) red[t] += red[t + s];
    __syncthreads();
  }
  float den = red[0] + 1e-6f;
  if (t < 64) {
    float a = 0.f;
    const float* m = Ms2 + (size_t)c * 8192;
    const float* q = cq + c * 128;
#pragma unroll 4
    for (int i = 0; i < 128; ++i) a += q[i] * m[i * 64 + t];
    outc[c * 64 + t] = a / den;
  }
}

__global__ void __launch_bounds__(64) k_final(
    const float* __restrict__ outc, const float* __restrict__ Wc1,
    const float* __restrict__ bc1, const float* __restrict__ Wc2,
    const float* __restrict__ bc2, float* __restrict__ logits)
{
  __shared__ float g[64];
  __shared__ float hid[32];
  int bg = blockIdx.x, t = threadIdx.x;
  float a = 0.f;
  for (int p = 0; p < PP; ++p) a += outc[((bg * PP + p) * 64) + t];
  g[t] = a * (1.f / PP);
  __syncthreads();
  if (t < 32) {
    float h = bc1[t];
    for (int d = 0; d < 64; ++d) h += g[d] * Wc1[d * 32 + t];
    hid[t] = h > 0.f ? h : 0.f;
  }
  __syncthreads();
  if (t < NCC) {
    float l = bc2[t];
    for (int k = 0; k < 32; ++k) l += hid[k] * Wc2[k * NCC + t];
    logits[bg * NCC + t] = l;
  }
}

// ---------------------------------------------------------------------------
extern "C" void kernel_launch(void* const* d_in, const int* in_sizes, int n_in,
                              void* d_out, int out_size, void* d_ws, size_t ws_size,
                              hipStream_t stream) {
  const float* x      = (const float*)d_in[0];
  const int*  mapper  = (const int*)d_in[1];
  const int*  sb      = (const int*)d_in[2];
  const int*  erow    = (const int*)d_in[3];
  const int*  ecol    = (const int*)d_in[4];
  const float* attr   = (const float*)d_in[5];
  const float* W_in1  = (const float*)d_in[6];  const float* b_in1  = (const float*)d_in[7];
  const float* W_in2  = (const float*)d_in[8];  const float* b_in2  = (const float*)d_in[9];
  const float* W_pre1 = (const float*)d_in[10]; const float* b_pre1 = (const float*)d_in[11];
  const float* W_pre2 = (const float*)d_in[12]; const float* b_pre2 = (const float*)d_in[13];
  const float* W_post1= (const float*)d_in[14]; const float* b_post1= (const float*)d_in[15];
  const float* W_post2= (const float*)d_in[16]; const float* b_post2= (const float*)d_in[17];
  const float* W_K    = (const float*)d_in[18]; const float* b_K    = (const float*)d_in[19];
  const float* W_V    = (const float*)d_in[20]; const float* b_V    = (const float*)d_in[21];
  const float* alphab = (const float*)d_in[22];
  const float* W_c1   = (const float*)d_in[23]; const float* b_c1   = (const float*)d_in[24];
  const float* W_c2   = (const float*)d_in[25]; const float* b_c2   = (const float*)d_in[26];
  float* logits = (float*)d_out;

  // --- workspace carve (256B aligned) ---
  char* w = (char*)d_ws;
  auto alloc = [&](size_t bytes) {
    char* p = w;
    w += (bytes + 255) & ~(size_t)255;
    return p;
  };
  _Float16* x16  = (_Float16*)alloc((size_t)NN * 64 * 2);
  _Float16* h16  = (_Float16*)alloc((size_t)NN * 64 * 2);
  _Float16* sx16 = (_Float16*)alloc((size_t)SS * 64 * 2);
  _Float16* Wt_in1  = (_Float16*)alloc(8192);
  _Float16* Wt_in2  = (_Float16*)alloc(8192);
  _Float16* Wt_pre1 = (_Float16*)alloc(8192);
  _Float16* Wt_pre2 = (_Float16*)alloc(8192);
  _Float16* Wt_post1= (_Float16*)alloc(8192);
  _Float16* Wt_post2= (_Float16*)alloc(8192);
  _Float16* Wt_K    = (_Float16*)alloc(8192);
  _Float16* Wt_V    = (_Float16*)alloc(8192);
  _Float16* Q16 = (_Float16*)alloc((size_t)SS * 64 * 2);
  _Float16* K16 = (_Float16*)alloc((size_t)SS * 64 * 2);
  _Float16* V16 = (_Float16*)alloc((size_t)SS * 64 * 2);
  float* Qs    = (float*)alloc((size_t)CC * 64 * 4);
  _Float16* Qs16 = (_Float16*)alloc((size_t)CC * 64 * 2);
  float* Qs2   = (float*)alloc((size_t)CC * 64 * 4);
  float* Ksum  = (float*)alloc((size_t)CC * 64 * 4);
  float* Ksk   = (float*)alloc((size_t)CC * 64 * 4);
  int* cnts    = (int*)alloc(CC * 4);
  int* offs    = (int*)alloc(CC * 4);
  float* deg   = (float*)alloc(CC * 4);
  float* ea    = (float*)alloc(EE * 4);
  float* sab   = (float*)alloc(256);
  float* cq    = (float*)alloc((size_t)CC * 128 * 4);
  _Float16* ck16 = (_Float16*)alloc((size_t)SS * 128 * 2);
  float* ck_s  = (float*)alloc((size_t)CC * 128 * 4);
  float* cks2  = (float*)alloc((size_t)CC * 128 * 4);
  float* Ms    = (float*)alloc((size_t)CC * 8192 * 4);
  float* Ms2   = (float*)alloc((size_t)CC * 8192 * 4);
  float* outc  = (float*)alloc((size_t)CC * 64 * 4);

  // --- zero accumulators (graph-capture-legal memset nodes) ---
  hipMemsetAsync(Qs,   0, (size_t)CC * 64 * 4, stream);
  hipMemsetAsync(Ksum, 0, (size_t)CC * 64 * 4, stream);
  hipMemsetAsync(ck_s, 0, (size_t)CC * 128 * 4, stream);
  hipMemsetAsync(cks2, 0, (size_t)CC * 128 * 4, stream);
  hipMemsetAsync(Ms2,  0, (size_t)CC * 8192 * 4, stream);
  hipMemsetAsync(deg,  0, CC * 4, stream);
  hipMemsetAsync(cnts, 0, CC * 4, stream);

  // --- weight transpose/convert (8 x 64x64) ---
  k_wt<<<16, 256, 0, stream>>>(W_in1,  Wt_in1);
  k_wt<<<16, 256, 0, stream>>>(W_in2,  Wt_in2);
  k_wt<<<16, 256, 0, stream>>>(W_pre1, Wt_pre1);
  k_wt<<<16, 256, 0, stream>>>(W_pre2, Wt_pre2);
  k_wt<<<16, 256, 0, stream>>>(W_post1,Wt_post1);
  k_wt<<<16, 256, 0, stream>>>(W_post2,Wt_post2);
  k_wt<<<16, 256, 0, stream>>>(W_K,    Wt_K);
  k_wt<<<16, 256, 0, stream>>>(W_V,    Wt_V);

  // --- input MLP (WMMA + TDM), gather, graph prep ---
  k_cvt<<<(NN * 64 + 255) / 256, 256, 0, stream>>>(x, x16, NN * 64);
  k_mlp2<<<NN / 128, 256, 0, stream>>>(x16, Wt_in1, b_in1, Wt_in2, b_in2,
                                       nullptr, h16, 1);
  k_gather<<<(SS * 8 + 255) / 256, 256, 0, stream>>>(h16, mapper, sx16);
  k_count<<<SS / 256, 256, 0, stream>>>(sb, cnts);
  k_deg<<<EE / 256, 256, 0, stream>>>(erow, attr, deg);
  k_scan<<<1, 1024, 0, stream>>>(cnts, offs);
  k_ea<<<EE / 256, 256, 0, stream>>>(erow, ecol, attr, deg, ea);

  // --- Q / K / V (WMMA + TDM) ---
  k_mlp2<<<SS / 128, 256, 0, stream>>>(sx16, Wt_pre1, b_pre1, Wt_pre2, b_pre2,
                                       nullptr, Q16, 0);
  k_gemm1<<<SS / 128, 256, 0, stream>>>(sx16, Wt_K, b_K, K16);
  k_gemm1<<<SS / 128, 256, 0, stream>>>(sx16, Wt_V, b_V, V16);
  k_segsumQK<<<(SS * 64) / 256, 256, 0, stream>>>(sb, Q16, K16, Qs, Ksum);

  // --- post deepset on Qs (WMMA + TDM) ---
  k_cvt<<<(CC * 64 + 255) / 256, 256, 0, stream>>>(Qs, Qs16, CC * 64);
  k_mlp2<<<CC / 128, 256, 0, stream>>>(Qs16, Wt_post1, b_post1, Wt_post2, b_post2,
                                       Qs2, nullptr, 0);

  // --- kernelized attention assembly ---
  k_alpha<<<1, 1, 0, stream>>>(alphab, sab);
  k_cqKs<<<(CC * 64) / 256, 256, 0, stream>>>(Ksum, cnts, Qs2, sab, Ksk, cq);
  k_ck<<<(SS * 64) / 256, 256, 0, stream>>>(sb, K16, Ksk, sab, ck16, ck_s);

  // --- per-cluster outer products (WMMA) + edge propagation ---
  k_ms<<<CC, 256, 0, stream>>>(ck16, V16, offs, cnts, Ms);
  k_prop<<<EE, 256, 0, stream>>>(erow, ecol, ea, Ms, ck_s, Ms2, cks2);

  // --- H/den, readout, classifier ---
  k_hout<<<CC, 128, 0, stream>>>(cq, cks2, Ms2, outc);
  k_final<<<BB, 64, 0, stream>>>(outc, W_c1, b_c1, W_c2, b_c2, logits);

  (void)in_sizes; (void)n_in; (void)out_size; (void)ws_size;
}
